// CGCNN_Interactions_78073915507063
// MI455X (gfx1250) — matrix operand order repaired and checked by reading, not compiled, not run on hardware
//
#include <hip/hip_runtime.h>

typedef __attribute__((ext_vector_type(16))) _Float16 v16h;
typedef __attribute__((ext_vector_type(8)))  _Float16 v8h;
typedef __attribute__((ext_vector_type(8)))  float    v8f;

#define NFILT   64
#define NGAUSS  5
#define NNODES  10000
#define NEDGES  50000
#define MTILES  (NEDGES/16)   // 3125 tiles of 16 edges
#define KTILES  130           // (65*64)/32 : z extended with 1.0 row folds in nn2_b
#define ZSTR    66            // padded z row stride (64 z + 1.0 + pad)

// ---------------------------------------------------------------------------
// Prep: rearrange nn2_w/nn2_b into exact WMMA B-fragment order.
// T'[k,o] = nn2_w[h, i*64+o] with k = h*64+i (h<64), = nn2_b[i*64+o] for h==64.
// Storage: [kt][nt][lane][16 halves], lane<16 holds K=kt*32+j, lane>=16 holds
// K=kt*32+16+j, n = nt*16 + lane%16. 532,480 bytes.
// ---------------------------------------------------------------------------
__global__ void k_build_bfrag(const float* __restrict__ nn2_w,
                              const float* __restrict__ nn2_b,
                              _Float16* __restrict__ bfrag) {
    int t = blockIdx.x * blockDim.x + threadIdx.x;
    if (t >= KTILES * 2048) return;
    int j    = t & 15;
    int lane = (t >> 4) & 31;
    int nt   = (t >> 9) & 3;
    int kt   = t >> 11;
    int kloc = ((lane & 16) ? 16 : 0) + j;
    int k    = kt * 32 + kloc;
    int n    = nt * 16 + (lane & 15);
    int hh   = k >> 6, ii = k & 63;
    float v = (hh < 64) ? nn2_w[hh * 4096 + ii * 64 + n] : nn2_b[ii * 64 + n];
    bfrag[t] = (_Float16)v;
}

// out0 = relu(h @ lin0_w + lin0_b)   [N,64] -- tiny, VALU
__global__ void k_lin0(const float* __restrict__ hin, const float* __restrict__ w,
                       const float* __restrict__ b, float* __restrict__ out) {
    int t = blockIdx.x * blockDim.x + threadIdx.x;
    if (t >= NNODES * 64) return;
    int v = t >> 6, o = t & 63;
    const float* hr = hin + (size_t)v * 64;
    float s = b[o];
#pragma unroll
    for (int i = 0; i < 64; ++i) s += hr[i] * w[i * 64 + o];
    out[t] = fmaxf(s, 0.f);
}

// z = relu(relu(edge_attr@short_w+short_b)@nn1_w+nn1_b), stored f16, row padded
// with a trailing 1.0 (bias row of the fused GEMM).
__global__ void k_edge_z(const float* __restrict__ edge_attr,
                         const float* __restrict__ short_w, const float* __restrict__ short_b,
                         const float* __restrict__ nn1_w,   const float* __restrict__ nn1_b,
                         _Float16* __restrict__ zext) {
    int t = blockIdx.x * blockDim.x + threadIdx.x;
    if (t >= NEDGES * 64) return;
    int e = t >> 6, hh = t & 63;
    const float* ar = edge_attr + (size_t)e * NGAUSS;
    float s = nn1_b[hh];
#pragma unroll
    for (int c = 0; c < 3; ++c) {
        float ea = short_b[c];
#pragma unroll
        for (int g = 0; g < NGAUSS; ++g) ea += ar[g] * short_w[g * 3 + c];
        ea = fmaxf(ea, 0.f);
        s += ea * nn1_w[c * 64 + hh];
    }
    s = fmaxf(s, 0.f);
    size_t base = (size_t)e * ZSTR;
    zext[base + hh] = (_Float16)s;
    if (hh == 0) { zext[base + 64] = (_Float16)1.0f; zext[base + 65] = (_Float16)0.0f; }
}

__global__ void k_zero(float* __restrict__ p, int n) {
    int t = blockIdx.x * blockDim.x + threadIdx.x;
    if (t < n) p[t] = 0.f;
}

__global__ void k_cnt(const int* __restrict__ dst, float* __restrict__ cnt) {
    int t = blockIdx.x * blockDim.x + threadIdx.x;
    if (t < NEDGES) unsafeAtomicAdd(&cnt[dst[t]], 1.0f);
}

__global__ void k_f32tof16(const float* __restrict__ s, _Float16* __restrict__ d, int n) {
    int t = blockIdx.x * blockDim.x + threadIdx.x;
    if (t < n) d[t] = (_Float16)s[t];
}

// ---------------------------------------------------------------------------
// Fused message GEMM: MSG[E,64] = (z⊗x)[E,4160] @ T'[4160,64], scatter-add to
// agg[dst]. One wave = 2 M-tiles (32 edges), branch-free (out-of-range tile
// reads edge 0, scatters into trash row NNODES). K-loop iterates h = 0..64;
// both 32-wide i-tiles of that h are consumed per iteration, all fragment
// indices compile-time constant. Loop kept unroll-1 with a sched barrier so
// the scheduler cannot rotate accumulator registers across iterations.
// ---------------------------------------------------------------------------
__global__ void __launch_bounds__(256)
k_conv(const _Float16* __restrict__ xh, const _Float16* __restrict__ zext,
       const _Float16* __restrict__ bfrag, const int* __restrict__ src,
       const int* __restrict__ dst, float* __restrict__ agg) {
    const int lane = threadIdx.x & 31;
    const int wid  = blockIdx.x * (blockDim.x >> 5) + (threadIdx.x >> 5);
    const int t0   = wid * 2;
    if (t0 >= MTILES) return;                 // wave-uniform
    const int  mrow = lane & 15;
    const int  hi   = (lane >> 4) & 1;        // lane half
    const int  boff = hi ? 16 : 0;

    v16h xf[2][2];
    const _Float16* zb[2];
    unsigned de[2][8];                        // 32-bit agg row indices

#pragma unroll
    for (int u = 0; u < 2; ++u) {
        const int  tl  = t0 + u;
        const bool pad = (tl >= MTILES);      // wave-uniform, select-only below
        const int  tc  = pad ? 0 : tl;        // clamped tile for loads
        const int  e   = tc * 16 + mrow;
        const _Float16* xr = xh + (size_t)src[e] * 64;
        // A-layout X fragments for i-tiles 0 and 1 (two b128 loads each)
        v8h p0 = *(const v8h*)(xr + boff);
        v8h p1 = *(const v8h*)(xr + boff + 8);
        v8h p2 = *(const v8h*)(xr + 32 + boff);
        v8h p3 = *(const v8h*)(xr + 32 + boff + 8);
        v16h f0, f1;
#pragma unroll
        for (int q = 0; q < 8; ++q) {
            f0[q] = p0[q]; f0[q + 8] = p1[q];
            f1[q] = p2[q]; f1[q + 8] = p3[q];
        }
        xf[u][0] = f0; xf[u][1] = f1;
        zb[u] = zext + (size_t)e * ZSTR;
#pragma unroll
        for (int r = 0; r < 8; ++r)
            de[u][r] = pad ? (unsigned)NNODES : (unsigned)dst[tc * 16 + r + hi * 8];
    }

    v8f acc[2][4];
#pragma unroll
    for (int u = 0; u < 2; ++u)
#pragma unroll
        for (int nt = 0; nt < 4; ++nt) acc[u][nt] = {};

    const _Float16* bpl = bfrag + (size_t)lane * 16;
#pragma clang loop unroll(disable)
    for (int hh = 0; hh < 65; ++hh) {
        __builtin_amdgcn_sched_barrier(0);    // no cross-iteration reg rotation
        // kt0 = 2*hh (i-tile 0), kt1 = 2*hh+1 (i-tile 1); kt stride = 2048 halves
        const _Float16* bp = bpl + (size_t)hh * 4096;
        const _Float16 z0 = zb[0][hh];
        const _Float16 z1 = zb[1][hh];
        v16h a00 = xf[0][0] * z0;             // 8x v_pk_mul_f16 each
        v16h a01 = xf[0][1] * z0;
        v16h a10 = xf[1][0] * z1;
        v16h a11 = xf[1][1] * z1;
#pragma unroll
        for (int nt = 0; nt < 4; ++nt) {
            v16h b0 = *(const v16h*)(bp + nt * 512);          // kt0, pre-swizzled
            v16h b1 = *(const v16h*)(bp + 2048 + nt * 512);   // kt1
            acc[0][nt] = __builtin_amdgcn_wmma_f32_16x16x32_f16(
                false, a00, false, b0, (short)0, acc[0][nt], false, false);
            acc[1][nt] = __builtin_amdgcn_wmma_f32_16x16x32_f16(
                false, a10, false, b0, (short)0, acc[1][nt], false, false);
            acc[0][nt] = __builtin_amdgcn_wmma_f32_16x16x32_f16(
                false, a01, false, b1, (short)0, acc[0][nt], false, false);
            acc[1][nt] = __builtin_amdgcn_wmma_f32_16x16x32_f16(
                false, a11, false, b1, (short)0, acc[1][nt], false, false);
        }
    }
    __builtin_amdgcn_sched_barrier(0);

    // scatter: acc element r on this lane = edge row (r + hi*8), col nt*16+mrow
#pragma unroll
    for (int u = 0; u < 2; ++u)
#pragma unroll
        for (int nt = 0; nt < 4; ++nt)
#pragma unroll
            for (int r = 0; r < 8; ++r)
                unsafeAtomicAdd(agg + (de[u][r] * 64u + (unsigned)(nt * 16 + mrow)),
                                acc[u][nt][r]);
}

// out' = relu(out @ root_w + agg/max(cnt,1) + conv_b)
__global__ void k_update(const float* __restrict__ outin, const float* __restrict__ agg,
                         const float* __restrict__ cnt, const float* __restrict__ root_w,
                         const float* __restrict__ conv_b, float* __restrict__ outnext) {
    int t = blockIdx.x * blockDim.x + threadIdx.x;
    if (t >= NNODES * 64) return;
    int v = t >> 6, o = t & 63;
    const float* r = outin + (size_t)v * 64;
    float s = conv_b[o] + agg[t] / fmaxf(cnt[v], 1.0f);
#pragma unroll
    for (int i = 0; i < 64; ++i) s += r[i] * root_w[i * 64 + o];
    outnext[t] = fmaxf(s, 0.f);
}

extern "C" void kernel_launch(void* const* d_in, const int* in_sizes, int n_in,
                              void* d_out, int out_size, void* d_ws, size_t ws_size,
                              hipStream_t stream) {
    const float* hin      = (const float*)d_in[0];
    const int*   ei       = (const int*)d_in[1];   // [2,E]
    const float* edge_attr= (const float*)d_in[3];
    const float* lin0_w   = (const float*)d_in[5];
    const float* lin0_b   = (const float*)d_in[6];
    const float* short_w  = (const float*)d_in[7];
    const float* short_b  = (const float*)d_in[8];
    const float* nn1_w    = (const float*)d_in[9];
    const float* nn1_b    = (const float*)d_in[10];
    const float* nn2_w    = (const float*)d_in[11];
    const float* nn2_b    = (const float*)d_in[12];
    const float* root_w   = (const float*)d_in[13];
    const float* conv_b   = (const float*)d_in[14];
    const int* src = ei;
    const int* dst = ei + NEDGES;
    (void)in_sizes; (void)n_in; (void)out_size; (void)ws_size;

    char* ws = (char*)d_ws;
    size_t off = 0;
    auto alloc = [&](size_t bytes) {
        void* p = ws + off;
        off = (off + bytes + 255) & ~(size_t)255;
        return p;
    };
    float*    out_a = (float*)alloc((size_t)NNODES * 64 * 4);
    float*    out_b = (float*)alloc((size_t)NNODES * 64 * 4);
    float*    agg   = (float*)alloc((size_t)(NNODES + 1) * 64 * 4);  // +1 trash row
    float*    cnt   = (float*)alloc((size_t)NNODES * 4);
    _Float16* bfrag = (_Float16*)alloc((size_t)KTILES * 2048 * 2);
    _Float16* zext  = (_Float16*)alloc((size_t)NEDGES * ZSTR * 2);
    _Float16* xh    = (_Float16*)alloc((size_t)NNODES * 64 * 2);

    const dim3 blk(256);
    k_build_bfrag<<<(KTILES * 2048 + 255) / 256, blk, 0, stream>>>(nn2_w, nn2_b, bfrag);
    k_lin0<<<(NNODES * 64 + 255) / 256, blk, 0, stream>>>(hin, lin0_w, lin0_b, out_a);
    k_edge_z<<<(NEDGES * 64 + 255) / 256, blk, 0, stream>>>(edge_attr, short_w, short_b,
                                                            nn1_w, nn1_b, zext);
    k_zero<<<(NNODES + 255) / 256, blk, 0, stream>>>(cnt, NNODES);
    k_cnt<<<(NEDGES + 255) / 256, blk, 0, stream>>>(dst, cnt);

    const int nWaves     = (MTILES + 1) / 2;       // 1563 waves, 2 M-tiles each
    const int convBlocks = (nWaves + 7) / 8;       // 8 waves / 256-thread block

    float* cur = out_a;
    for (int p = 0; p < 2; ++p) {
        k_f32tof16<<<(NNODES * 64 + 255) / 256, blk, 0, stream>>>(cur, xh, NNODES * 64);
        k_zero<<<(NNODES * 64 + 255) / 256, blk, 0, stream>>>(agg, NNODES * 64);
        k_conv<<<convBlocks, blk, 0, stream>>>(xh, zext, bfrag, src, dst, agg);
        float* nxt = (p == 0) ? out_b : (float*)d_out;
        k_update<<<(NNODES * 64 + 255) / 256, blk, 0, stream>>>(cur, agg, cnt,
                                                                root_w, conv_b, nxt);
        cur = nxt;
    }
}
